// vsta_layer_36867999268877
// MI455X (gfx1250) — matrix operand order repaired and checked by reading, not compiled, not run on hardware
//
#include <hip/hip_runtime.h>

typedef __attribute__((ext_vector_type(16))) __bf16 v16bf;
typedef __attribute__((ext_vector_type(8)))  float  v8f;

#define B_    4
#define T_    12
#define N_    325
#define D_    512
#define H_    8
#define HD_   64
#define WIN_  3
#define NT_   21            // ceil(325/16) key tiles
#define NPAD_ 336           // NT_*16
#define KP_   352           // key dim padded to 11*32 for A*V wmma
#define TW_   (T_*WIN_)     // 36
#define QKV_ELEMS ((size_t)B_*T_*H_*N_*HD_)      // 7,987,200 bf16 each
#define HCAT_ELEMS ((size_t)B_*TW_*N_*D_)        // 23,961,600 f32

// ---------------------------------------------------------------------------
// Kernel 1: y = x @ W^T + b, output as bf16 in (b,t,h,n,hd) layout.
// M = B*T*N = 15600 = 975*16 (exact), K = 512, Nout = 512.
// Block: 256 thr = 8 waves; each wave owns one 16x16 tile (16 rows x 128 cols
// per block). All WMMA fragments load as contiguous runs straight from global.
// ---------------------------------------------------------------------------
__global__ __launch_bounds__(256) void qkv_proj_kernel(
    const float* __restrict__ X, const float* __restrict__ W,
    const float* __restrict__ bias, __bf16* __restrict__ out)
{
  const int wave = threadIdx.x >> 5;
  const int lane = threadIdx.x & 31;
  const int l16  = lane & 15;
  const bool hi  = lane >= 16;
  const int rbase = blockIdx.x * 16;
  const int ecol  = blockIdx.y * 128 + wave * 16 + l16;

  const float* xrow = X + (size_t)(rbase + l16) * D_;   // A row (fixed per lane)
  const float* wrow = W + (size_t)ecol * D_;            // B col == W row (y = xW^T)

  v8f acc = {0.f, 0.f, 0.f, 0.f, 0.f, 0.f, 0.f, 0.f};
  for (int kb = 0; kb < D_; kb += 32) {
    v16bf a, bm;
    const int a0 = kb + (hi ? 8 : 0);          // A: two runs of 8 (K and K+16)
    #pragma unroll
    for (int i = 0; i < 8; ++i) {
      a[i]     = (__bf16)xrow[a0 + i];
      a[i + 8] = (__bf16)xrow[a0 + 16 + i];
    }
    const int b0 = kb + (hi ? 8 : 0);          // B: two runs of 8 (K and K+16)
    #pragma unroll
    for (int i = 0; i < 16; ++i) bm[i] = (__bf16)wrow[b0 + (i < 8 ? i : i + 8)];
    acc = __builtin_amdgcn_wmma_f32_16x16x32_bf16(false, a, false, bm,
                                                  (short)0, acc, false, false);
  }

  const float bb  = bias[ecol];
  const int  h    = ecol >> 6;
  const int  hdi  = ecol & 63;
  #pragma unroll
  for (int j = 0; j < 8; ++j) {
    const int m   = j + (hi ? 8 : 0);
    const int r   = rbase + m;
    const int b   = r / (T_ * N_);
    const int rem = r - b * (T_ * N_);
    const int t   = rem / N_;
    const int n   = rem - t * N_;
    out[(((size_t)((b * T_ + t) * H_ + h) * N_ + n) * HD_) + hdi] =
        (__bf16)(acc[j] + bb);
  }
}

// ---------------------------------------------------------------------------
// Kernel 2: windowed attention. Block = (qtile, t*win, b*h); 8 waves.
//   S = (q*scale^p) K^T ; S = (S+1)*adj ; A = softmax(S) ; R = A V / rowsum
// Scores staged in LDS (16x336 f32); probs as bf16 (16x352, zero padded).
// ---------------------------------------------------------------------------
__global__ __launch_bounds__(256) void attn_kernel(
    const __bf16* __restrict__ q, const __bf16* __restrict__ k,
    const __bf16* __restrict__ v, const float* __restrict__ adj,
    float* __restrict__ hcat)
{
  __shared__ float  sbuf[16][NPAD_];
  __shared__ __bf16 abuf[16][KP_];
  __shared__ float  red[16][16];
  __shared__ float  rowmax[16];
  __shared__ float  rinv[16];

  const int qt = blockIdx.x;
  const int ti = blockIdx.y / T_;
  const int t  = blockIdx.y % T_;
  const int b  = blockIdx.z >> 3;
  const int h  = blockIdx.z & 7;

  const int   koff_arr[3] = {3, 5, 6};                 // cumulative k/v shift
  const float qsc_arr[3]  = {0.125f, 0.015625f, 0.001953125f}; // scale^(ti+1)
  const int   tk  = (t + T_ - koff_arr[ti]) % T_;
  const int   ta  = (t + 9 + ti) % T_;                 // adj uses single shift
  const float qsc = qsc_arr[ti];

  const __bf16* qb = q + ((size_t)((b * T_ + t)  * H_ + h) * N_) * HD_;
  const __bf16* kb = k + ((size_t)((b * T_ + tk) * H_ + h) * N_) * HD_;
  const __bf16* vb = v + ((size_t)((b * T_ + tk) * H_ + h) * N_) * HD_;
  const float*  ab = adj + ((size_t)(b * T_ + ta) * N_) * N_;

  const int wave = threadIdx.x >> 5;
  const int lane = threadIdx.x & 31;
  const int l16  = lane & 15;
  const bool hi  = lane >= 16;

  // ---- phase 1: score tiles (waves split the 21 key tiles) ----
  const int qrow = qt * 16 + l16;
  const int qrc  = qrow < N_ ? qrow : N_ - 1;
  const __bf16* qrp = qb + (size_t)qrc * HD_;
  v16bf qa[2];
  #pragma unroll
  for (int c = 0; c < 2; ++c) {
    const int a0 = c * 32 + (hi ? 8 : 0);
    #pragma unroll
    for (int i = 0; i < 8; ++i) {
      qa[c][i]     = qrp[a0 + i];
      qa[c][i + 8] = qrp[a0 + 16 + i];
    }
  }
  for (int j = wave; j < NT_; j += 8) {
    const int nk  = j * 16 + l16;
    const int nkc = nk < N_ ? nk : N_ - 1;
    const __bf16* krp = kb + (size_t)nkc * HD_;
    v8f acc = {0.f, 0.f, 0.f, 0.f, 0.f, 0.f, 0.f, 0.f};
    #pragma unroll
    for (int c = 0; c < 2; ++c) {
      v16bf bfrag;
      const int b0 = c * 32 + (hi ? 8 : 0);
      #pragma unroll
      for (int i = 0; i < 16; ++i) bfrag[i] = krp[b0 + (i < 8 ? i : i + 8)];
      acc = __builtin_amdgcn_wmma_f32_16x16x32_bf16(false, qa[c], false, bfrag,
                                                    (short)0, acc, false, false);
    }
    const int coln = j * 16 + l16;
    #pragma unroll
    for (int jj = 0; jj < 8; ++jj) {
      const int m  = jj + (hi ? 8 : 0);
      const int nq = qt * 16 + m;
      float s = -1.0e30f;
      if (nq < N_ && coln < N_)
        s = (acc[jj] * qsc + 1.0f) * ab[(size_t)nq * N_ + coln];
      sbuf[m][coln] = s;
    }
  }
  __syncthreads();

  // ---- phase 2: row softmax (16 threads per row) ----
  {
    const int row = threadIdx.x >> 4;
    const int c16 = threadIdx.x & 15;
    float mx = -3.0e38f;
    for (int col = c16; col < N_; col += 16) mx = fmaxf(mx, sbuf[row][col]);
    red[row][c16] = mx;
    __syncthreads();
    if (c16 == 0) {
      float m2 = red[row][0];
      #pragma unroll
      for (int i = 1; i < 16; ++i) m2 = fmaxf(m2, red[row][i]);
      rowmax[row] = m2;
    }
    __syncthreads();
    const float rm = rowmax[row];
    float sum = 0.f;
    for (int col = c16; col < KP_; col += 16) {
      float p = 0.f;
      if (col < N_) p = __expf(sbuf[row][col] - rm);
      abuf[row][col] = (__bf16)p;
      sum += p;
    }
    red[row][c16] = sum;
    __syncthreads();
    if (c16 == 0) {
      float s2 = 0.f;
      #pragma unroll
      for (int i = 0; i < 16; ++i) s2 += red[row][i];
      rinv[row] = 1.0f / s2;
    }
    __syncthreads();
  }

  // ---- phase 3: R = A * V (waves 0..3 each own 16 head-dim cols) ----
  if (wave < 4) {
    const int hdcol = wave * 16 + l16;
    v8f acc = {0.f, 0.f, 0.f, 0.f, 0.f, 0.f, 0.f, 0.f};
    for (int kk = 0; kk < KP_; kk += 32) {
      v16bf afrag, bfrag;
      const int a0 = kk + (hi ? 8 : 0);
      #pragma unroll
      for (int i = 0; i < 8; ++i) {
        afrag[i]     = abuf[l16][a0 + i];
        afrag[i + 8] = abuf[l16][a0 + 16 + i];
      }
      const int b0 = kk + (hi ? 8 : 0);
      #pragma unroll
      for (int i = 0; i < 16; ++i) {
        int kr = b0 + (i < 8 ? i : i + 8);
        kr = kr < N_ ? kr : N_ - 1;          // padded A entries are zero
        bfrag[i] = vb[(size_t)kr * HD_ + hdcol];
      }
      acc = __builtin_amdgcn_wmma_f32_16x16x32_bf16(false, afrag, false, bfrag,
                                                    (short)0, acc, false, false);
    }
    const int wt = ti * T_ + t;
    #pragma unroll
    for (int jj = 0; jj < 8; ++jj) {
      const int m  = jj + (hi ? 8 : 0);
      const int nq = qt * 16 + m;
      if (nq < N_) {
        hcat[((size_t)(b * TW_ + wt) * N_ + nq) * D_ + h * HD_ + hdcol] =
            acc[jj] * rinv[m];
      }
    }
  }
}

// ---------------------------------------------------------------------------
// Kernel 3: temporal mix (36 -> 12) + bias + residual + LayerNorm.
// Block per (n, b); 36x512 h-slab staged through LDS in two 256-col halves.
// ---------------------------------------------------------------------------
__global__ __launch_bounds__(256) void mix_ln_kernel(
    const float* __restrict__ hcat, const float* __restrict__ Wd,
    const float* __restrict__ bd, const float* __restrict__ X,
    const float* __restrict__ gamma, const float* __restrict__ beta,
    float* __restrict__ out)
{
  __shared__ float hl[TW_ * 256];     // 36 KB half-slab
  __shared__ float wd[T_ * TW_];      // 12x36
  __shared__ float redk[256];
  __shared__ float s_mu, s_iv;

  const int n   = blockIdx.x;
  const int b   = blockIdx.y;
  const int tid = threadIdx.x;

  for (int i = tid; i < T_ * TW_; i += 256) wd[i] = Wd[i];

  float yv[2][T_];
  for (int half = 0; half < 2; ++half) {
    const int dbase = half * 256;
    __syncthreads();                                  // done reading prev half
    for (int idx = tid; idx < TW_ * 256; idx += 256) {
      const int w  = idx >> 8;
      const int dd = idx & 255;
      hl[idx] = hcat[((size_t)(b * TW_ + w) * N_ + n) * D_ + dbase + dd];
    }
    __syncthreads();
    const int d = dbase + tid;
    #pragma unroll
    for (int t = 0; t < T_; ++t) {
      float acc = bd[t];
      #pragma unroll
      for (int w = 0; w < TW_; ++w) acc += hl[w * 256 + tid] * wd[t * TW_ + w];
      acc += X[((size_t)(b * T_ + t) * N_ + n) * D_ + d];
      yv[half][t] = acc;
    }
  }

  const float g0 = gamma[tid], g1 = gamma[tid + 256];
  const float e0 = beta[tid],  e1 = beta[tid + 256];

  #pragma unroll
  for (int t = 0; t < T_; ++t) {
    redk[tid] = yv[0][t] + yv[1][t];
    __syncthreads();
    for (int s = 128; s > 0; s >>= 1) {
      if (tid < s) redk[tid] += redk[tid + s];
      __syncthreads();
    }
    if (tid == 0) s_mu = redk[0] * (1.0f / 512.0f);
    __syncthreads();
    const float mu = s_mu;
    const float d0 = yv[0][t] - mu, d1 = yv[1][t] - mu;
    redk[tid] = d0 * d0 + d1 * d1;
    __syncthreads();
    for (int s = 128; s > 0; s >>= 1) {
      if (tid < s) redk[tid] += redk[tid + s];
      __syncthreads();
    }
    if (tid == 0) s_iv = rsqrtf(redk[0] * (1.0f / 512.0f) + 1e-5f);
    __syncthreads();
    const float iv = s_iv;
    const size_t ob = ((size_t)(b * T_ + t) * N_ + n) * D_;
    out[ob + tid]       = d0 * iv * g0 + e0;
    out[ob + tid + 256] = d1 * iv * g1 + e1;
    __syncthreads();
  }
}

// ---------------------------------------------------------------------------
extern "C" void kernel_launch(void* const* d_in, const int* in_sizes, int n_in,
                              void* d_out, int out_size, void* d_ws, size_t ws_size,
                              hipStream_t stream) {
  const float* x     = (const float*)d_in[0];
  const float* adj   = (const float*)d_in[1];
  // d_in[2] = s_adj (unused by the reference)
  const float* Wq    = (const float*)d_in[3];
  const float* bq    = (const float*)d_in[4];
  const float* Wk    = (const float*)d_in[5];
  const float* bk    = (const float*)d_in[6];
  const float* Wv    = (const float*)d_in[7];
  const float* bv    = (const float*)d_in[8];
  const float* Wd    = (const float*)d_in[9];
  const float* bd    = (const float*)d_in[10];
  const float* gamma = (const float*)d_in[11];
  const float* beta  = (const float*)d_in[12];
  float* out = (float*)d_out;

  __bf16* qws = (__bf16*)d_ws;
  __bf16* kws = qws + QKV_ELEMS;
  __bf16* vws = kws + QKV_ELEMS;
  float*  hcat = (float*)(vws + QKV_ELEMS);

  const dim3 g1(975, 4);                       // 15600/16 x (512/128)
  qkv_proj_kernel<<<g1, 256, 0, stream>>>(x, Wq, bq, qws);
  qkv_proj_kernel<<<g1, 256, 0, stream>>>(x, Wk, bk, kws);
  qkv_proj_kernel<<<g1, 256, 0, stream>>>(x, Wv, bv, vws);

  attn_kernel<<<dim3(NT_, T_ * WIN_, B_ * H_), 256, 0, stream>>>(
      qws, kws, vws, adj, hcat);

  mix_ln_kernel<<<dim3(N_, B_), 256, 0, stream>>>(
      hcat, Wd, bd, x, gamma, beta, out);
}
